// BDH_66056597013022
// MI455X (gfx1250) — matrix operand order repaired and checked by reading, not compiled, not run on hardware
//
#include <hip/hip_runtime.h>
#include <hip/hip_bf16.h>

// ---------------------------------------------------------------------------
// Problem constants (match reference)
// ---------------------------------------------------------------------------
#define CB   4        // batch
#define CT   1024     // seq len
#define CH   4        // heads
#define CN   4096     // N (H*Dh)
#define CD   256      // model dim
#define CL   6        // layers
#define CV   256      // vocab
#define CDH  1024     // per-head dim
#define CEPS 1e-5f

#define USE_ASYNC_LDS 1   // CDNA5 global_load_async_to_lds_b128 staging path

typedef __attribute__((ext_vector_type(16))) __bf16 v16bf;
typedef __attribute__((ext_vector_type(8)))  float  v8f;

// ---------------------------------------------------------------------------
// bf16 helpers (round-to-nearest-even), stored as u16 in buffers
// ---------------------------------------------------------------------------
__device__ __forceinline__ unsigned short f2bf(float f) {
    unsigned u = __float_as_uint(f);
    u += 0x7FFFu + ((u >> 16) & 1u);
    return (unsigned short)(u >> 16);
}
__device__ __forceinline__ float bf2f(unsigned short h) {
    return __uint_as_float(((unsigned)h) << 16);
}

// ---------------------------------------------------------------------------
// CDNA5 async copy: memory -> LDS, 16B per lane, tracked by ASYNCcnt
// ---------------------------------------------------------------------------
__device__ __forceinline__ void async_cp_b128(unsigned ldsOff, const void* g) {
#if USE_ASYNC_LDS
    asm volatile("global_load_async_to_lds_b128 %0, %1, off"
                 :: "v"(ldsOff), "v"(g) : "memory");
#endif
}
#define WAIT_ASYNC6() asm volatile("s_wait_asynccnt 6" ::: "memory")
#define WAIT_ASYNC0() asm volatile("s_wait_asynccnt 0" ::: "memory")

// LDS byte offset of a __shared__ element (flat-addr low 32 bits == LDS addr)
#define LDS_OFF(p) ((unsigned)(unsigned long long)(p))

// ---------------------------------------------------------------------------
// Generic batched bf16 WMMA GEMM: C = epi(A @ Bt^T)
//   A : row-major [M x K], lda          (bf16)
//   Bt: row-major [N x K], ldb          (bf16)  -- "transposed B" only
//   block = 128 threads (4 waves), tile 64(M) x 128(N), K step 32
//   each wave: 32x64 patch = 2x4 v_wmma_f32_16x16x32_bf16 tiles
//   double-buffered LDS, async staged
// EPI: 0 = bf16 store, 1 = relu->bf16, 2 = relu*aux->bf16, 3 = f32 store
// ---------------------------------------------------------------------------
struct GemmParams {
    const unsigned short* __restrict__ A;
    const unsigned short* __restrict__ B;   // Bt [N x K]
    unsigned short* __restrict__ Cb;
    float* __restrict__ Cf;
    const unsigned short* __restrict__ aux; // X for EPI==2
    int K;
    int lda, ldb, ldc, ldaux;
    int inner;
    long long sA_o, sA_i, sB_o, sB_i, sC_o, sC_i, sX_o, sX_i;
};

#define LP 48   // LDS row pitch in bf16 elements (96B: 16B-aligned, bank-skewed)

template<int EPI>
__launch_bounds__(128)
__global__ void bdh_gemm(GemmParams p) {
    __shared__ unsigned short LdsA[2][64 * LP];
    __shared__ unsigned short LdsB[2][128 * LP];

    const int z  = blockIdx.z;
    const int zo = z / p.inner;
    const int zi = z - zo * p.inner;
    const unsigned short* Abase = p.A + zo * p.sA_o + zi * p.sA_i;
    const unsigned short* Bbase = p.B + zo * p.sB_o + zi * p.sB_i;
    const long long cOff = zo * p.sC_o + zi * p.sC_i;
    const long long xOff = zo * p.sX_o + zi * p.sX_i;

    const int tid  = threadIdx.x;
    const int lane = tid & 31;
    const int wave = tid >> 5;
    const int wr   = wave >> 1;            // wave M half (0..1)
    const int wc   = wave & 1;             // wave N half (0..1), x64
    const int m0   = blockIdx.y * 64;
    const int n0   = blockIdx.x * 128;
    const int hi   = lane >> 4;
    const int lr   = lane & 15;

    // per-thread staging coordinates: one uint4 (8 bf16) chunk per issue
    const int arow = tid >> 2;             // 0..31
    const int acol = (tid & 3) * 8;        // 0,8,16,24
    const unsigned short* aSrc = Abase + (long long)(m0 + arow) * p.lda + acol;
    const unsigned short* bSrc = Bbase + (long long)(n0 + arow) * p.ldb + acol;
    const long long aRowStep = (long long)32 * p.lda;
    const long long bRowStep = (long long)32 * p.ldb;

    v8f acc[2][4] = {};

#if USE_ASYNC_LDS
    auto stage = [&](int buf) {
        async_cp_b128(LDS_OFF(&LdsA[buf][(arow     ) * LP + acol]), aSrc);
        async_cp_b128(LDS_OFF(&LdsA[buf][(arow + 32) * LP + acol]), aSrc + aRowStep);
        #pragma unroll
        for (int i = 0; i < 4; ++i)
            async_cp_b128(LDS_OFF(&LdsB[buf][(arow + i * 32) * LP + acol]),
                          bSrc + i * bRowStep);
    };
    stage(0);
    aSrc += 32; bSrc += 32;
#endif

    int buf = 0;
    for (int k0 = 0; k0 < p.K; k0 += 32) {
#if USE_ASYNC_LDS
        if (k0 + 32 < p.K) {
            stage(buf ^ 1);
            aSrc += 32; bSrc += 32;
            WAIT_ASYNC6();                 // previous tile's 6 copies done
        } else {
            WAIT_ASYNC0();
        }
        __syncthreads();
#else
        {   // synchronous fallback: regs -> LDS
            uint4 a0 = *(const uint4*)(aSrc);
            uint4 a1 = *(const uint4*)(aSrc + aRowStep);
            uint4 b0 = *(const uint4*)(bSrc);
            uint4 b1 = *(const uint4*)(bSrc + bRowStep);
            uint4 b2 = *(const uint4*)(bSrc + 2 * bRowStep);
            uint4 b3 = *(const uint4*)(bSrc + 3 * bRowStep);
            *(uint4*)(&LdsA[0][(arow     ) * LP + acol]) = a0;
            *(uint4*)(&LdsA[0][(arow + 32) * LP + acol]) = a1;
            *(uint4*)(&LdsB[0][(arow      ) * LP + acol]) = b0;
            *(uint4*)(&LdsB[0][(arow + 32 ) * LP + acol]) = b1;
            *(uint4*)(&LdsB[0][(arow + 64 ) * LP + acol]) = b2;
            *(uint4*)(&LdsB[0][(arow + 96 ) * LP + acol]) = b3;
            aSrc += 32; bSrc += 32;
        }
        __syncthreads();
#endif
        // ---- fragments per documented 16-bit VGPR layouts ----
        const unsigned short* As = LdsA[buf];
        const unsigned short* Bs = LdsB[buf];
        v16bf af[2], bfv[4];
        #pragma unroll
        for (int mt = 0; mt < 2; ++mt) {
            const int m = wr * 32 + mt * 16 + lr;
            union { uint4 u[2]; v16bf v; } cv;
            cv.u[0] = *(const uint4*)(&As[m * LP + hi * 8]);      // K = hi*8 + 0..7
            cv.u[1] = *(const uint4*)(&As[m * LP + 16 + hi * 8]); // K = 16+hi*8+0..7
            af[mt] = cv.v;
        }
        #pragma unroll
        for (int nt = 0; nt < 4; ++nt) {
            const int n = wc * 64 + nt * 16 + lr;
            union { uint4 u[2]; v16bf v; } cv;
            cv.u[0] = *(const uint4*)(&Bs[n * LP + hi * 8]);      // K = hi*8 + 0..7
            cv.u[1] = *(const uint4*)(&Bs[n * LP + 16 + hi * 8]); // K = 16+hi*8+0..7
            bfv[nt] = cv.v;
        }
        #pragma unroll
        for (int mt = 0; mt < 2; ++mt)
            #pragma unroll
            for (int nt = 0; nt < 4; ++nt)
                acc[mt][nt] = __builtin_amdgcn_wmma_f32_16x16x32_bf16(
                    false, af[mt], false, bfv[nt],
                    (short)0, acc[mt][nt], false, false);
        __syncthreads();
        buf ^= 1;
    }

    // ---- epilogue; f32 C/D layout: VGPR r, lane l -> M=(l>>4)*8+r, N=l&15 ----
#if !USE_ASYNC_LDS
    buf = 0;
#endif
    #pragma unroll
    for (int mt = 0; mt < 2; ++mt) {
        #pragma unroll
        for (int nt = 0; nt < 4; ++nt) {
            const int gn = n0 + wc * 64 + nt * 16 + lr;
            #pragma unroll
            for (int r = 0; r < 8; ++r) {
                const int gm = m0 + wr * 32 + mt * 16 + hi * 8 + r;
                float val = acc[mt][nt][r];
                if (EPI == 1) {
                    val = fmaxf(val, 0.f);
                } else if (EPI == 2) {
                    val = fmaxf(val, 0.f) *
                          bf2f(p.aux[xOff + (long long)gm * p.ldaux + gn]);
                }
                if (EPI == 3)
                    p.Cf[cOff + (long long)gm * p.ldc + gn] = val;
                else
                    p.Cb[cOff + (long long)gm * p.ldc + gn] = f2bf(val);
            }
        }
    }
}

// ---------------------------------------------------------------------------
// Batched transpose + fp32 -> bf16:  in [z][R][C] f32  ->  out [z][C][R] bf16
// ---------------------------------------------------------------------------
__global__ void bdh_transpose_cvt(const float* __restrict__ in,
                                  unsigned short* __restrict__ out,
                                  int R, int C, int total) {
    int idx = blockIdx.x * blockDim.x + threadIdx.x;
    if (idx >= total) return;
    int rc = R * C;
    int zz = idx / rc;
    int rem = idx - zz * rc;
    int r = rem / C;
    int c = rem - r * C;
    out[(long long)zz * rc + (long long)c * R + r] = f2bf(in[idx]);
}

// ---------------------------------------------------------------------------
// Block reduction helper (256 threads)
// ---------------------------------------------------------------------------
__device__ __forceinline__ float block_sum_256(float v, float* red, int d) {
    red[d] = v;
    __syncthreads();
    #pragma unroll
    for (int s = 128; s > 0; s >>= 1) {
        if (d < s) red[d] += red[d + s];
        __syncthreads();
    }
    float r = red[0];
    __syncthreads();
    return r;
}

// ---------------------------------------------------------------------------
// v = LN(emb_w[tokens]); writes fp32, bf16 and transposed-bf16 copies.
// 1 block per (b,t) row, 256 threads.
// ---------------------------------------------------------------------------
__global__ void bdh_embed_ln(const int* __restrict__ tokens,
                             const float* __restrict__ emb_w,
                             float* __restrict__ v,
                             unsigned short* __restrict__ vb,
                             unsigned short* __restrict__ vbT) {
    __shared__ float red[CD];
    const int row = blockIdx.x;
    const int d   = threadIdx.x;
    const int tok = tokens[row];
    float x = emb_w[(long long)tok * CD + d];
    const float mean = block_sum_256(x, red, d) * (1.f / CD);
    const float dx   = x - mean;
    const float var  = block_sum_256(dx * dx, red, d) * (1.f / CD);
    const float o    = dx * rsqrtf(var + CEPS);
    const int b = row / CT, t = row - b * CT;
    v[(long long)row * CD + d]  = o;
    vb[(long long)row * CD + d] = f2bf(o);
    vbT[(long long)b * CD * CT + (long long)d * CT + t] = f2bf(o);
}

// ---------------------------------------------------------------------------
// RoPE on X [B,H,T,Dh] -> Xr.  Pairs (e, e+Dh/2), cos/sin computed on the fly.
// ---------------------------------------------------------------------------
__global__ void bdh_rope(const unsigned short* __restrict__ X,
                         unsigned short* __restrict__ Xr) {
    const long long idx = (long long)blockIdx.x * blockDim.x + threadIdx.x;
    const long long total = (long long)CB * CH * CT * (CDH / 2);
    if (idx >= total) return;
    const int i   = (int)(idx & (CDH / 2 - 1));
    const int t   = (int)((idx >> 9) & (CT - 1));
    const long long zz = idx >> 19;
    const long long base = zz * (long long)CT * CDH + (long long)t * CDH;
    const float inv = __expf(-__logf(10000.f) * (2.f * (float)i) * (1.f / (float)CDH));
    float s, c;
    __sincosf((float)t * inv, &s, &c);
    const float x1 = bf2f(X[base + i]);
    const float x2 = bf2f(X[base + i + CDH / 2]);
    Xr[base + i]           = f2bf(x1 * c - x2 * s);
    Xr[base + i + CDH / 2] = f2bf(x2 * c + x1 * s);
}

// ---------------------------------------------------------------------------
// v = LN(v + LN(Z));  writes fp32, bf16, transposed-bf16.  1 block per row.
// ---------------------------------------------------------------------------
__global__ void bdh_ln_residual(const float* __restrict__ Z,
                                float* __restrict__ v,
                                unsigned short* __restrict__ vb,
                                unsigned short* __restrict__ vbT) {
    __shared__ float red[CD];
    const int row = blockIdx.x;
    const int d   = threadIdx.x;
    const long long o = (long long)row * CD + d;

    float z = Z[o];
    float m1 = block_sum_256(z, red, d) * (1.f / CD);
    float dz = z - m1;
    float v1 = block_sum_256(dz * dz, red, d) * (1.f / CD);
    float g  = dz * rsqrtf(v1 + CEPS);

    float t  = v[o] + g;
    float m2 = block_sum_256(t, red, d) * (1.f / CD);
    float dt = t - m2;
    float v2 = block_sum_256(dt * dt, red, d) * (1.f / CD);
    float out = dt * rsqrtf(v2 + CEPS);

    const int b = row / CT, tt = row - b * CT;
    v[o]  = out;
    vb[o] = f2bf(out);
    vbT[(long long)b * CD * CT + (long long)d * CT + tt] = f2bf(out);
}

// ---------------------------------------------------------------------------
// Host launcher
// ---------------------------------------------------------------------------
template<int EPI>
static inline void launch_gemm(const GemmParams& p, int gx, int gy, int gz,
                               hipStream_t s) {
    bdh_gemm<EPI><<<dim3(gx, gy, gz), 128, 0, s>>>(p);
}

extern "C" void kernel_launch(void* const* d_in, const int* in_sizes, int n_in,
                              void* d_out, int out_size, void* d_ws, size_t ws_size,
                              hipStream_t stream) {
    (void)in_sizes; (void)n_in; (void)out_size; (void)ws_size;
    const int*   tokens  = (const int*)d_in[0];
    const float* emb_w   = (const float*)d_in[1];
    const float* E       = (const float*)d_in[2];
    const float* Dx      = (const float*)d_in[3];
    const float* Dy      = (const float*)d_in[4];
    const float* readout = (const float*)d_in[5];
    float* out = (float*)d_out;

    char* ws = (char*)d_ws;
    size_t off = 0;
    auto carve = [&](size_t bytes) -> char* {
        char* p = ws + off;
        off += (bytes + 255) & ~(size_t)255;
        return p;
    };
    float*          v_f32 = (float*)         carve((size_t)CB * CT * CD * 4);
    unsigned short* v_bf  = (unsigned short*)carve((size_t)CB * CT * CD * 2);
    unsigned short* vT_bf = (unsigned short*)carve((size_t)CB * CD * CT * 2);
    unsigned short* DxT   = (unsigned short*)carve((size_t)CH * CDH * CD * 2);
    unsigned short* DyT   = (unsigned short*)carve((size_t)CH * CDH * CD * 2);
    unsigned short* ET    = (unsigned short*)carve((size_t)CD * CN * 2);
    unsigned short* roT   = (unsigned short*)carve((size_t)CV * CD * 2);
    unsigned short* X_bf  = (unsigned short*)carve((size_t)CB * CH * CT * CDH * 2);
    unsigned short* Xr_bf = (unsigned short*)carve((size_t)CB * CH * CT * CDH * 2);
    unsigned short* S_bf  = (unsigned short*)carve((size_t)CB * CH * CT * CT * 2);
    unsigned short* A_bf  = (unsigned short*)carve((size_t)CB * CH * CT * CD * 2);
    unsigned short* Y_bf  = (unsigned short*)carve((size_t)CB * CT * CN * 2);
    float*          Z_f32 = (float*)         carve((size_t)CB * CT * CD * 4);

    // Weight transposes (fp32 -> bf16, [z][R][C] -> [z][C][R])
    {
        int n1 = CH * CD * CDH;  // Dx/Dy: [H][D][Dh] -> [H][Dh][D]
        bdh_transpose_cvt<<<(n1 + 255) / 256, 256, 0, stream>>>(Dx, DxT, CD, CDH, n1);
        bdh_transpose_cvt<<<(n1 + 255) / 256, 256, 0, stream>>>(Dy, DyT, CD, CDH, n1);
        int n2 = CN * CD;        // E: [N][D] -> [D][N]
        bdh_transpose_cvt<<<(n2 + 255) / 256, 256, 0, stream>>>(E, ET, CN, CD, n2);
        int n3 = CD * CV;        // readout: [D][V] -> [V][D]
        bdh_transpose_cvt<<<(n3 + 255) / 256, 256, 0, stream>>>(readout, roT, CD, CV, n3);
    }

    // Embedding + LayerNorm (also produces v^T)
    bdh_embed_ln<<<CB * CT, CD, 0, stream>>>(tokens, emb_w, v_f32, v_bf, vT_bf);

    const long long TD  = (long long)CT * CD;
    const long long TDH = (long long)CT * CDH;
    const long long TT  = (long long)CT * CT;
    const long long TN  = (long long)CT * CN;
    const long long DDH = (long long)CD * CDH;

    for (int l = 0; l < CL; ++l) {
        // (a) X = relu(v @ Dx[h]) : M=T, N=Dh, K=D, z = b*H + h
        {
            GemmParams p = {};
            p.A = v_bf; p.B = DxT; p.Cb = X_bf;
            p.K = CD;
            p.lda = CD; p.ldb = CD; p.ldc = CDH;
            p.inner = CH;
            p.sA_o = TD;  p.sA_i = 0;
            p.sB_o = 0;   p.sB_i = DDH;
            p.sC_o = (long long)CH * TDH; p.sC_i = TDH;
            launch_gemm<1>(p, CDH / 128, CT / 64, CB * CH, stream);
        }
        // (b) RoPE
        {
            long long tot = (long long)CB * CH * CT * (CDH / 2);
            bdh_rope<<<(unsigned)((tot + 255) / 256), 256, 0, stream>>>(X_bf, Xr_bf);
        }
        // (c) S = Xr @ Xr^T : M=N=T, K=Dh  (Bt == Xr, naturally [N x K])
        {
            GemmParams p = {};
            p.A = Xr_bf; p.B = Xr_bf; p.Cb = S_bf;
            p.K = CDH;
            p.lda = CDH; p.ldb = CDH; p.ldc = CT;
            p.inner = 1;
            p.sA_o = TDH; p.sB_o = TDH; p.sC_o = TT;
            launch_gemm<0>(p, CT / 128, CT / 64, CB * CH, stream);
        }
        // (d) A = S @ v : M=T, N=D, K=T  (Bt = v^T [D x T] per batch)
        {
            GemmParams p = {};
            p.A = S_bf; p.B = vT_bf; p.Cb = A_bf;
            p.K = CT;
            p.lda = CT; p.ldb = CT; p.ldc = CD;
            p.inner = CH;
            p.sA_o = (long long)CH * TT; p.sA_i = TT;
            p.sB_o = (long long)CD * CT; p.sB_i = 0;
            p.sC_o = (long long)CH * TD; p.sC_i = TD;
            launch_gemm<0>(p, CD / 128, CT / 64, CB * CH, stream);
        }
        // (e) Y = relu(A @ Dy[h]) * X -> [B,T,H*Dh] interleaved
        {
            GemmParams p = {};
            p.A = A_bf; p.B = DyT; p.Cb = Y_bf; p.aux = X_bf;
            p.K = CD;
            p.lda = CD; p.ldb = CD; p.ldc = CN; p.ldaux = CDH;
            p.inner = CH;
            p.sA_o = (long long)CH * TD; p.sA_i = TD;
            p.sB_o = 0; p.sB_i = DDH;
            p.sC_o = TN; p.sC_i = CDH;
            p.sX_o = (long long)CH * TDH; p.sX_i = TDH;
            launch_gemm<2>(p, CDH / 128, CT / 64, CB * CH, stream);
        }
        // (f) Z = Y @ E : M=B*T, N=D, K=N  (Bt = E^T [D x N])
        {
            GemmParams p = {};
            p.A = Y_bf; p.B = ET; p.Cf = Z_f32;
            p.K = CN;
            p.lda = CN; p.ldb = CN; p.ldc = CD;
            p.inner = 1;
            launch_gemm<3>(p, CD / 128, (CB * CT) / 64, 1, stream);
        }
        // (g) v = LN(v + LN(Z))  (also refreshes v^T)
        bdh_ln_residual<<<CB * CT, CD, 0, stream>>>(Z_f32, v_f32, v_bf, vT_bf);
    }

    // Readout: out = v @ readout : M=B*T, N=V, K=D  (Bt = readout^T [V x D])
    {
        GemmParams p = {};
        p.A = v_bf; p.B = roT; p.Cf = out;
        p.K = CD;
        p.lda = CD; p.ldb = CD; p.ldc = CV;
        p.inner = 1;
        launch_gemm<3>(p, CV / 128, (CB * CT) / 64, 1, stream);
    }
}
